// SpatialBlock_24532853194965
// MI455X (gfx1250) — compile-verified, matrix operand order checked by reading, not run on hardware
//
#include <hip/hip_runtime.h>
#include <hip/hip_bf16.h>
#include <hip/hip_fp16.h>

typedef __attribute__((ext_vector_type(16))) _Float16 v16h;
typedef __attribute__((ext_vector_type(8)))  _Float16 v8h;
typedef __attribute__((ext_vector_type(8)))  float    v8f;
typedef __attribute__((ext_vector_type(4)))  unsigned int u32x4;
typedef __attribute__((ext_vector_type(8)))  int      i32x8;
typedef __attribute__((ext_vector_type(4)))  int      i32x4;

#if __has_builtin(__builtin_amdgcn_tensor_load_to_lds)
#define HAVE_TDM 1
#else
#define HAVE_TDM 0
#endif

__device__ __forceinline__ void wait_tensorcnt0() {
#if __has_builtin(__builtin_amdgcn_s_wait_tensorcnt)
  __builtin_amdgcn_s_wait_tensorcnt(0);
#else
  asm volatile("s_wait_tensorcnt 0x0" ::: "memory");
#endif
}

// ---------------------------------------------------------------------------
// WMMA 16x32 f16 fragment load, row-major source (stride ld halves).
// A layout (ISA 7.12.2): lane r=l&15 holds row r; K 0..7/16..23 (lanes 0-15),
// K 8..15/24..31 (lanes 16-31).  Two 16-byte loads.  Same layout serves the
// B operand of D = A*W^T for row-major W (N x K): lane = output col = W row.
// Works for both global and LDS sources.
// ---------------------------------------------------------------------------
__device__ __forceinline__ v16h load_frag(const _Float16* base, int ld) {
  const int lane = threadIdx.x & 31;
  const int r    = lane & 15;
  const int hi   = lane >> 4;
  const _Float16* rp = base + (size_t)r * ld + hi * 8;
  v8h lo = *(const v8h*)rp;
  v8h hh = *(const v8h*)(rp + 16);
  v16h out;
#pragma unroll
  for (int i = 0; i < 8; ++i) { out[i] = lo[i]; out[i + 8] = hh[i]; }
  return out;
}

#define WMMA_F16(A, B, C) \
  __builtin_amdgcn_wmma_f32_16x16x32_f16(false, (A), false, (B), (short)0, (C), false, false)

// ---------------------------------------------------------------------------
// Stage a 64-row x K-half weight tile into LDS via the Tensor Data Mover.
// TDM pad feature inserts 4 DWORDs (8 halves) after each K-half row, giving
// LDS row stride K+8 so fragment reads avoid bank conflicts.
// Issued by wave 0 only; TENSORcnt drained before the workgroup barrier.
// ---------------------------------------------------------------------------
template <int K>
__device__ __forceinline__ void stage_w_tile(_Float16* WS, const _Float16* Wg) {
#if HAVE_TDM
  if (threadIdx.x < 32) {
    const unsigned long long ga = (unsigned long long)(uintptr_t)Wg;
    const unsigned lds_off = (unsigned)(uintptr_t)WS;  // low 32 bits = LDS offset
    u32x4 g0;
    g0[0] = 1u;                                        // count=1, user mode
    g0[1] = lds_off;                                   // lds_addr (bytes)
    g0[2] = (unsigned)(ga & 0xffffffffu);              // global_addr[31:0]
    g0[3] = (unsigned)((ga >> 32) & 0x01ffffffu) | (2u << 30);  // addr[56:32] | type=2
    const int pad_iv = (K == 512) ? 7 : 6;             // interval = 2^(c+1) DWORDs = row
    i32x8 g1;
    g1[0] = (1 << 16) | (1 << 20) | (pad_iv << 22) | (3 << 25);  // 2B elems, pad 4 dw
    g1[1] = (K & 0xffff) << 16;                        // tensor_dim0 lo (ab_addr=0)
    g1[2] = (64 << 16);                                // dim0 hi=0 | tensor_dim1=64
    g1[3] = ((K & 0xffff) << 16);                      // dim1 hi=0 | tile_dim0=K
    g1[4] = 64;                                        // tile_dim1=64, tile_dim2=0
    g1[5] = K;                                         // tensor_dim0_stride lo
    g1[6] = 0;
    g1[7] = 0;
    i32x4 z4 = {0, 0, 0, 0};                           // groups 2/3 unused (2D)
    i32x8 z8 = {0, 0, 0, 0, 0, 0, 0, 0};
    __builtin_amdgcn_tensor_load_to_lds(g0, g1, z4, z4, z8, 0);
    wait_tensorcnt0();
  }
#else
  const int ldl = K + 8;
  for (int idx = threadIdx.x; idx < 64 * (K / 8); idx += 256) {
    const int r = idx / (K / 8), c = (idx % (K / 8)) * 8;
    *(v8h*)&WS[r * ldl + c] = *(const v8h*)&Wg[r * K + c];
  }
#endif
  __syncthreads();
}

// ---------------------------------------------------------------------------
// Wave tile 32x64, K unrolled by 2 with ping-pong A-fragments:
// even K-step uses (a0,a1) while (c0,c1) loads; odd step uses (c0,c1) while
// (a0,a1) reloads.  No register rotation -> no v_mov blocks, no WMMA->VALU
// WAR hazard nops, no per-iteration branch.  The final prefetch over-reads
// <=64 halves past the tile, which stays inside the workspace by layout.
// ---------------------------------------------------------------------------
template <int K>
__device__ __forceinline__ void gemm_core(const _Float16* A, int ldA,
                                          const _Float16* WS, size_t m0, v8f acc[8]) {
  const int ldl = K + 8;
  const _Float16* A0 = A + m0 * ldA;
  const _Float16* A1 = A + (m0 + 16) * ldA;
  v16h a0 = load_frag(A0, ldA);
  v16h a1 = load_frag(A1, ldA);
#pragma unroll 1
  for (int k0 = 0; k0 < K; k0 += 64) {
    v16h c0 = load_frag(A0 + k0 + 32, ldA);   // prefetch odd step
    v16h c1 = load_frag(A1 + k0 + 32, ldA);
    {
      v16h b0 = load_frag(WS + 0 * 16 * ldl + k0, ldl);
      v16h b1 = load_frag(WS + 1 * 16 * ldl + k0, ldl);
      v16h b2 = load_frag(WS + 2 * 16 * ldl + k0, ldl);
      v16h b3 = load_frag(WS + 3 * 16 * ldl + k0, ldl);
      acc[0] = WMMA_F16(a0, b0, acc[0]);
      acc[1] = WMMA_F16(a0, b1, acc[1]);
      acc[2] = WMMA_F16(a0, b2, acc[2]);
      acc[3] = WMMA_F16(a0, b3, acc[3]);
      acc[4] = WMMA_F16(a1, b0, acc[4]);
      acc[5] = WMMA_F16(a1, b1, acc[5]);
      acc[6] = WMMA_F16(a1, b2, acc[6]);
      acc[7] = WMMA_F16(a1, b3, acc[7]);
    }
    a0 = load_frag(A0 + k0 + 64, ldA);        // prefetch next even step
    a1 = load_frag(A1 + k0 + 64, ldA);        // (harmless overread on last)
    {
      const int k1 = k0 + 32;
      v16h b0 = load_frag(WS + 0 * 16 * ldl + k1, ldl);
      v16h b1 = load_frag(WS + 1 * 16 * ldl + k1, ldl);
      v16h b2 = load_frag(WS + 2 * 16 * ldl + k1, ldl);
      v16h b3 = load_frag(WS + 3 * 16 * ldl + k1, ldl);
      acc[0] = WMMA_F16(c0, b0, acc[0]);
      acc[1] = WMMA_F16(c0, b1, acc[1]);
      acc[2] = WMMA_F16(c0, b2, acc[2]);
      acc[3] = WMMA_F16(c0, b3, acc[3]);
      acc[4] = WMMA_F16(c1, b0, acc[4]);
      acc[5] = WMMA_F16(c1, b1, acc[5]);
      acc[6] = WMMA_F16(c1, b2, acc[6]);
      acc[7] = WMMA_F16(c1, b3, acc[7]);
    }
  }
}

// ---------------------------------------------------------------------------
// f32 -> f16 weight conversion
// ---------------------------------------------------------------------------
__global__ void k_f32tof16(const float* __restrict__ src, _Float16* __restrict__ dst, int n) {
  int i = blockIdx.x * 256 + threadIdx.x;
  if (i < n) dst[i] = (_Float16)src[i];
}

// ---------------------------------------------------------------------------
// LayerNorm (one wave per token, 8 ch/lane), optional fused shift+partition.
// ---------------------------------------------------------------------------
__global__ void k_ln(const float* __restrict__ x, const float* __restrict__ g,
                     const float* __restrict__ b, _Float16* __restrict__ dst,
                     int shifted) {
  const int wave = threadIdx.x >> 5;
  const int lane = threadIdx.x & 31;
  const size_t token = (size_t)blockIdx.x * 8 + wave;
  const float* xp = x + token * 256 + lane * 8;
  float v[8];
  float s1 = 0.f, s2 = 0.f;
#pragma unroll
  for (int i = 0; i < 8; ++i) { v[i] = xp[i]; s1 += v[i]; s2 += v[i] * v[i]; }
#pragma unroll
  for (int off = 16; off >= 1; off >>= 1) {
    s1 += __shfl_xor(s1, off, 32);
    s2 += __shfl_xor(s2, off, 32);
  }
  const float mu   = s1 * (1.f / 256.f);
  const float rstd = rsqrtf(s2 * (1.f / 256.f) - mu * mu + 1e-5f);
  size_t drow;
  if (shifted) {
    const int bb = (int)(token >> 12);
    const int n  = (int)(token & 4095);
    const int hs  = ((n >> 6) - 4) & 63;   // roll -4
    const int ws2 = ((n & 63) - 4) & 63;
    drow = (size_t)(bb * 64 + (hs >> 3) * 8 + (ws2 >> 3)) * 64 + (hs & 7) * 8 + (ws2 & 7);
  } else {
    drow = token;
  }
  _Float16* dp = dst + drow * 256 + lane * 8;
#pragma unroll
  for (int i = 0; i < 8; ++i) {
    const int ch = lane * 8 + i;
    dp[i] = (_Float16)((v[i] - mu) * rstd * g[ch] + b[ch]);
  }
}

// ---------------------------------------------------------------------------
// GEMM kernels: 256 threads = 8 waves; block tile 256 rows x 64 cols.
// Weight tile staged in LDS by the TDM.
// ---------------------------------------------------------------------------
__global__ void k_gemm_qkv(const _Float16* __restrict__ A, const _Float16* __restrict__ W,
                           const float* __restrict__ bias, _Float16* __restrict__ out) {
  __shared__ __align__(16) _Float16 WS[64 * (256 + 8)];
  const int n0 = blockIdx.y * 64;
  stage_w_tile<256>(WS, W + (size_t)n0 * 256);
  const size_t m0 = (size_t)blockIdx.x * 256 + (threadIdx.x >> 5) * 32;
  v8f acc[8] = {};
  gemm_core<256>(A, 256, WS, m0, acc);
  const int lane = threadIdx.x & 31;
  const int rb = (lane >> 4) * 8, cl = lane & 15;
#pragma unroll
  for (int hf = 0; hf < 2; ++hf)
#pragma unroll
    for (int t = 0; t < 4; ++t) {
      const int col = n0 + 16 * t + cl;
      const float bv = bias[col];
#pragma unroll
      for (int gg = 0; gg < 8; ++gg)
        out[(m0 + hf * 16 + rb + gg) * 768 + col] = (_Float16)(acc[hf * 4 + t][gg] + bv);
    }
}

__global__ void k_gemm_proj(const _Float16* __restrict__ A, const _Float16* __restrict__ W,
                            const float* __restrict__ bias, const float* __restrict__ x,
                            const float* __restrict__ ls1, float* __restrict__ x1) {
  __shared__ __align__(16) _Float16 WS[64 * (256 + 8)];
  const int n0 = blockIdx.y * 64;
  stage_w_tile<256>(WS, W + (size_t)n0 * 256);
  const size_t m0 = (size_t)blockIdx.x * 256 + (threadIdx.x >> 5) * 32;
  v8f acc[8] = {};
  gemm_core<256>(A, 256, WS, m0, acc);
  const int lane = threadIdx.x & 31;
  const int rb = (lane >> 4) * 8, cl = lane & 15;
#pragma unroll
  for (int hf = 0; hf < 2; ++hf)
#pragma unroll
    for (int t = 0; t < 4; ++t) {
      const int col = n0 + 16 * t + cl;
      const float bv = bias[col], lv = ls1[col];
#pragma unroll
      for (int gg = 0; gg < 8; ++gg) {
        const size_t row = m0 + hf * 16 + rb + gg;   // windowed row
        const int wid = (int)(row >> 6), i = (int)(row & 63);
        const int bb = wid >> 6, widx = wid & 63;
        const int h = ((widx >> 3) * 8 + (i >> 3) + 4) & 63;  // reverse + roll +4
        const int w = ((widx & 7) * 8 + (i & 7) + 4) & 63;
        const size_t dst = ((size_t)bb * 4096 + h * 64 + w) * 256 + col;
        x1[dst] = x[dst] + lv * (acc[hf * 4 + t][gg] + bv);
      }
    }
}

__global__ void k_gemm_fc1(const _Float16* __restrict__ A, const _Float16* __restrict__ W,
                           const float* __restrict__ bias, _Float16* __restrict__ out) {
  __shared__ __align__(16) _Float16 WS[64 * (256 + 8)];
  const int n0 = blockIdx.y * 64;
  stage_w_tile<256>(WS, W + (size_t)n0 * 256);
  const size_t m0 = (size_t)blockIdx.x * 256 + (threadIdx.x >> 5) * 32;
  v8f acc[8] = {};
  gemm_core<256>(A, 256, WS, m0, acc);
  const int lane = threadIdx.x & 31;
  const int rb = (lane >> 4) * 8, cl = lane & 15;
#pragma unroll
  for (int hf = 0; hf < 2; ++hf)
#pragma unroll
    for (int t = 0; t < 4; ++t) {
      const int col = n0 + 16 * t + cl;
      const float bv = bias[col];
#pragma unroll
      for (int gg = 0; gg < 8; ++gg)
        out[(m0 + hf * 16 + rb + gg) * 512 + col] = (_Float16)(acc[hf * 4 + t][gg] + bv);
    }
}

__global__ void k_gemm_fc2(const _Float16* __restrict__ A, const _Float16* __restrict__ W,
                           const float* __restrict__ bias, const float* __restrict__ ls2,
                           float* __restrict__ out /* x1, updated in place */) {
  __shared__ __align__(16) _Float16 WS[64 * (512 + 8)];
  const int n0 = blockIdx.y * 64;
  stage_w_tile<512>(WS, W + (size_t)n0 * 512);
  const size_t m0 = (size_t)blockIdx.x * 256 + (threadIdx.x >> 5) * 32;
  v8f acc[8] = {};
  gemm_core<512>(A, 512, WS, m0, acc);
  const int lane = threadIdx.x & 31;
  const int rb = (lane >> 4) * 8, cl = lane & 15;
#pragma unroll
  for (int hf = 0; hf < 2; ++hf)
#pragma unroll
    for (int t = 0; t < 4; ++t) {
      const int col = n0 + 16 * t + cl;
      const float bv = bias[col], lv = ls2[col];
#pragma unroll
      for (int gg = 0; gg < 8; ++gg) {
        const size_t dst = (m0 + hf * 16 + rb + gg) * 256 + col;
        out[dst] = out[dst] + lv * (acc[hf * 4 + t][gg] + bv);  // x1 + ls2*f
      }
    }
}

// ---------------------------------------------------------------------------
// Windowed attention: one 128-thread workgroup per (window, head).
// ---------------------------------------------------------------------------
__global__ void k_attn(const _Float16* __restrict__ qkv, _Float16* __restrict__ out) {
  __shared__ __align__(16) float    S[64 * 64];
  __shared__ __align__(16) _Float16 P[64 * 64];
  __shared__ __align__(16) _Float16 Vl[64 * 32];
  const int wid  = blockIdx.x >> 3;
  const int head = blockIdx.x & 7;
  const int tid  = threadIdx.x;
  const int wave = tid >> 5, lane = tid & 31;
  const size_t rowbase = (size_t)wid * 64;

  { // stage V tile (64x32 f16) into LDS
    const int j = tid >> 1, dh = (tid & 1) * 16;
    const _Float16* src = qkv + (rowbase + j) * 768 + 512 + head * 32 + dh;
    *(v8h*)&Vl[j * 32 + dh]     = *(const v8h*)src;
    *(v8h*)&Vl[j * 32 + dh + 8] = *(const v8h*)(src + 8);
  }

  // S strip: rows 16*wave..+15, all 64 key columns (K = head_dim = 32)
  v16h aq = load_frag(qkv + (rowbase + wave * 16) * 768 + head * 32, 768);
  v16h bk[4];
#pragma unroll
  for (int t = 0; t < 4; ++t)
    bk[t] = load_frag(qkv + (rowbase + t * 16) * 768 + 256 + head * 32, 768);
  v8f sacc[4] = {};
#pragma unroll
  for (int t = 0; t < 4; ++t) sacc[t] = WMMA_F16(aq, bk[t], sacc[t]);

  const int rb = (lane >> 4) * 8, cl = lane & 15;
#pragma unroll
  for (int t = 0; t < 4; ++t)
#pragma unroll
    for (int gg = 0; gg < 8; ++gg)
      S[(wave * 16 + rb + gg) * 64 + t * 16 + cl] = sacc[t][gg];
  __syncthreads();

  if (tid < 64) {  // one thread per query row: ALiBi + shift mask + softmax
    const int i = tid;
    const int widx = wid & 63;
    const int hs_i = (widx >> 3) * 8 + (i >> 3);
    const int ws_i = (widx & 7) * 8 + (i & 7);
    const int gi = (hs_i < 56 ? 0 : (hs_i < 60 ? 1 : 2)) * 3 +
                   (ws_i < 56 ? 0 : (ws_i < 60 ? 1 : 2));
    const float slope = exp2f(-(float)(head + 1));
    const float scale = 0.17677669529663687f;  // 32^-0.5
    float mx = -1e30f;
#pragma unroll 1
    for (int j = 0; j < 64; ++j) {
      const int hs_j = (widx >> 3) * 8 + (j >> 3);
      const int ws_j = (widx & 7) * 8 + (j & 7);
      const int gj = (hs_j < 56 ? 0 : (hs_j < 60 ? 1 : 2)) * 3 +
                     (ws_j < 56 ? 0 : (ws_j < 60 ? 1 : 2));
      const float val = S[i * 64 + j] * scale - slope * fabsf((float)(i - j)) +
                        ((gi != gj) ? -100.f : 0.f);
      S[i * 64 + j] = val;
      mx = fmaxf(mx, val);
    }
    float sum = 0.f;
#pragma unroll 1
    for (int j = 0; j < 64; ++j) {
      const float e = __expf(S[i * 64 + j] - mx);
      S[i * 64 + j] = e;
      sum += e;
    }
    const float inv = 1.f / sum;
#pragma unroll 1
    for (int j = 0; j < 64; ++j) P[i * 64 + j] = (_Float16)(S[i * 64 + j] * inv);
  }
  __syncthreads();

  // O strip = P(16x64) * V(64x32)
  v8f oacc[2] = {};
#pragma unroll
  for (int kk = 0; kk < 64; kk += 32) {
    v16h ap = load_frag(P + wave * 16 * 64 + kk, 64);
#pragma unroll
    for (int t = 0; t < 2; ++t) {
      const int d = t * 16 + cl;
      const int klo = kk + (lane >> 4) * 8;
      v16h bv;
#pragma unroll
      for (int e = 0; e < 8; ++e) {
        bv[e]     = Vl[(klo + e) * 32 + d];
        bv[e + 8] = Vl[(klo + 16 + e) * 32 + d];
      }
      oacc[t] = WMMA_F16(ap, bv, oacc[t]);
    }
  }
#pragma unroll
  for (int t = 0; t < 2; ++t)
#pragma unroll
    for (int gg = 0; gg < 8; ++gg)
      out[(rowbase + wave * 16 + rb + gg) * 256 + head * 32 + t * 16 + cl] =
          (_Float16)oacc[t][gg];
}

// ---------------------------------------------------------------------------
// Depthwise 3x3 'SAME' conv + bias + exact GELU
// ---------------------------------------------------------------------------
__global__ void k_dwconv_gelu(const _Float16* __restrict__ hsrc,
                              const float* __restrict__ wgt,
                              const float* __restrict__ bias,
                              _Float16* __restrict__ gout) {
  const size_t idx = (size_t)blockIdx.x * 256 + threadIdx.x;
  const int c = (int)(idx & 511);
  const size_t p = idx >> 9;
  const int w = (int)(p & 63);
  const int h = (int)((p >> 6) & 63);
  const size_t bb = p >> 12;
  float acc = bias[c];
#pragma unroll
  for (int kh = -1; kh <= 1; ++kh) {
    const int hh = h + kh;
    if (hh < 0 || hh > 63) continue;
#pragma unroll
    for (int kw = -1; kw <= 1; ++kw) {
      const int ww = w + kw;
      if (ww < 0 || ww > 63) continue;
      acc += (float)hsrc[(bb * 4096 + hh * 64 + ww) * 512 + c] *
             wgt[((kh + 1) * 3 + (kw + 1)) * 512 + c];
    }
  }
  gout[idx] = (_Float16)(0.5f * acc * (1.f + erff(acc * 0.70710678118654752f)));
}

// ---------------------------------------------------------------------------
extern "C" void kernel_launch(void* const* d_in, const int* in_sizes, int n_in,
                              void* d_out, int out_size, void* d_ws, size_t ws_size,
                              hipStream_t stream) {
  (void)in_sizes; (void)n_in; (void)out_size; (void)ws_size;
  const float* x      = (const float*)d_in[0];
  const float* n1g    = (const float*)d_in[1];
  const float* n1b    = (const float*)d_in[2];
  const float* qkv_w  = (const float*)d_in[3];
  const float* qkv_b  = (const float*)d_in[4];
  const float* proj_w = (const float*)d_in[5];
  const float* proj_b = (const float*)d_in[6];
  const float* n2g    = (const float*)d_in[7];
  const float* n2b    = (const float*)d_in[8];
  const float* fc1_w  = (const float*)d_in[9];
  const float* fc1_b  = (const float*)d_in[10];
  const float* dw_w   = (const float*)d_in[11];
  const float* dw_b   = (const float*)d_in[12];
  const float* fc2_w  = (const float*)d_in[13];
  const float* fc2_b  = (const float*)d_in[14];
  const float* ls1    = (const float*)d_in[15];
  const float* ls2    = (const float*)d_in[16];
  float* out = (float*)d_out;
  char*  ws  = (char*)d_ws;

  const size_t MB = 1024ull * 1024ull;
  // region reuse (stream-ordered): yw/t2 [0,64MB) -> gbuf [0,128MB)
  // attn [64,128MB);  qkv [128,320MB) -> hbuf [128,256MB);  weights @320MB
  _Float16* yw   = (_Float16*)(ws);              // 131072 x 256
  _Float16* attn = (_Float16*)(ws + 64 * MB);    // 131072 x 256
  _Float16* gbuf = (_Float16*)(ws);              // 131072 x 512
  _Float16* qkvb = (_Float16*)(ws + 128 * MB);   // 131072 x 768
  _Float16* hbuf = (_Float16*)(ws + 128 * MB);   // 131072 x 512
  _Float16* w16     = (_Float16*)(ws + 320 * MB);
  _Float16* qkvw16  = w16;                       // 768 x 256
  _Float16* projw16 = w16 + 196608;              // 256 x 256
  _Float16* fc1w16  = w16 + 262144;              // 512 x 256
  _Float16* fc2w16  = w16 + 393216;              // 256 x 512

  k_f32tof16<<<dim3(768), 256, 0, stream>>>(qkv_w, qkvw16, 196608);
  k_f32tof16<<<dim3(256), 256, 0, stream>>>(proj_w, projw16, 65536);
  k_f32tof16<<<dim3(512), 256, 0, stream>>>(fc1_w, fc1w16, 131072);
  k_f32tof16<<<dim3(512), 256, 0, stream>>>(fc2_w, fc2w16, 131072);

  // LN1 + cyclic shift + window partition
  k_ln<<<dim3(16384), 256, 0, stream>>>(x, n1g, n1b, yw, 1);
  // qkv projection (131072x256 @ 256x768)
  k_gemm_qkv<<<dim3(512, 12), 256, 0, stream>>>(yw, qkvw16, qkv_b, qkvb);
  // attention per (window, head)
  k_attn<<<dim3(16384), 128, 0, stream>>>(qkvb, attn);
  // proj + window reverse + roll + residual*ls1 -> x1 (in d_out)
  k_gemm_proj<<<dim3(512, 4), 256, 0, stream>>>(attn, projw16, proj_b, x, ls1, out);
  // LN2
  k_ln<<<dim3(16384), 256, 0, stream>>>(out, n2g, n2b, yw, 0);
  // fc1 (131072x256 @ 256x512)
  k_gemm_fc1<<<dim3(512, 8), 256, 0, stream>>>(yw, fc1w16, fc1_b, hbuf);
  // depthwise 3x3 + GELU
  k_dwconv_gelu<<<dim3(262144), 256, 0, stream>>>(hbuf, dw_w, dw_b, gbuf);
  // fc2 + residual*ls2, in place on d_out (x1)
  k_gemm_fc2<<<dim3(512, 4), 256, 0, stream>>>(gbuf, fc2w16, fc2_b, ls2, out);
}
